// SelfAttention_70300024700965
// MI455X (gfx1250) — matrix operand order repaired and checked by reading, not compiled, not run on hardware
//
#include <hip/hip_runtime.h>

// ---------------------------------------------------------------------------
// CDNA5 (gfx1250) wave32 WMMA self-attention.
//  - All matmuls via v_wmma_f32_16x16x32_bf16 (f32 accumulate).
//  - One-time f32->bf16 (RNE) conversion; all staging is pure async copy via
//    GLOBAL_LOAD_ASYNC_TO_LDS_B128 with DOUBLE BUFFERING: async copies for
//    tile i+1 are in flight while tile i is consumed (in-order ASYNCcnt
//    completion => s_wait_asynccnt N waits exactly for the previous tile).
//  - Flash-style online-softmax attention, causal mask computed arithmetically.
// ---------------------------------------------------------------------------

typedef __attribute__((ext_vector_type(16))) __bf16 v16bf;
typedef __attribute__((ext_vector_type(8)))  float  v8f;
typedef __attribute__((ext_vector_type(4)))  unsigned int u32x4;
typedef __attribute__((ext_vector_type(2)))  float  f32x2;

union FragU { unsigned int u[8]; u32x4 q[2]; v16bf f; };

__device__ __forceinline__ unsigned int pack_bf16(float a, float b) {
  unsigned int ua = __float_as_uint(a);
  unsigned int ub = __float_as_uint(b);
  ua = (ua + 0x7FFFu + ((ua >> 16) & 1u)) >> 16;
  ub = (ub + 0x7FFFu + ((ub >> 16) & 1u)) >> 16;
  return (ua & 0xFFFFu) | (ub << 16);
}

__device__ __forceinline__ unsigned short f32_to_bf16(float a) {
  unsigned int ua = __float_as_uint(a);
  return (unsigned short)((ua + 0x7FFFu + ((ua >> 16) & 1u)) >> 16);
}

__device__ __forceinline__ v8f wmma_bf16(v16bf a, v16bf b, v8f c) {
  return __builtin_amdgcn_wmma_f32_16x16x32_bf16(false, a, false, b, (short)0, c,
                                                 false, false);
}

// Async 16B copy global->LDS (ASYNCcnt). lds generic pointer: addr[31:0] is
// the LDS byte offset (flat->LDS aperture mapping truncates high bits).
__device__ __forceinline__ void async_cp16(const void* g, void* lds) {
  unsigned loff = (unsigned)(unsigned long long)lds;
  asm volatile("global_load_async_to_lds_b128 %0, %1, off"
               :: "v"(loff), "v"(g) : "memory");
}
#define ASYNC_WAIT(n) asm volatile("s_wait_asynccnt " #n ::: "memory")

// A-matrix fragment (16x32 bf16) from LDS laid out [row][... k contiguous]:
// within the 64B k-window: chunks at (half*16) and (32 + half*16).
__device__ __forceinline__ v16bf frag_a_lds(const char* base, int rowBytes,
                                            int row, int kByteOff, int lane) {
  const char* p = base + row * rowBytes + kByteOff + ((lane >> 4) * 16);
  FragU fu;
  fu.q[0] = *(const u32x4*)(p);
  fu.q[1] = *(const u32x4*)(p + 32);
  return fu.f;
}

// B-matrix fragment (32x16 bf16) from LDS laid out [n][k contiguous]:
// within the 64B k-window: chunks at (half*16) and (32 + half*16).
__device__ __forceinline__ v16bf frag_b_lds(const char* base, int rowBytes,
                                            int row, int kByteOff, int lane) {
  const char* p = base + row * rowBytes + kByteOff + ((lane >> 4) * 16);
  FragU fu;
  fu.q[0] = *(const u32x4*)(p);
  fu.q[1] = *(const u32x4*)(p + 32);
  return fu.f;
}

// A fragment straight from global bf16 (row-major, rowStride elements).
__device__ __forceinline__ v16bf frag_a_global_bf16(const unsigned short* A,
                                                    int rowStride, int row0,
                                                    int col0, int lane) {
  int m = lane & 15, half = lane >> 4;
  const unsigned* rp =
      (const unsigned*)(A + (long)(row0 + m) * rowStride + col0);
  FragU fu;
#pragma unroll
  for (int d = 0; d < 8; ++d) {
    int c = (d < 4) ? (d + half * 4) : (4 + d + half * 4);
    fu.u[d] = rp[c];
  }
  return fu.f;
}

// 16-lane (half-wave) reductions via ds_swizzle (xor 1,2,4,8 stay inside each
// 16-lane half, matching WMMA C-layout row placement).
#define SWZF(v, imm) \
  __int_as_float(__builtin_amdgcn_ds_swizzle(__float_as_int(v), imm))

__device__ __forceinline__ float redmax16(float v) {
  v = fmaxf(v, SWZF(v, 0x041F));
  v = fmaxf(v, SWZF(v, 0x081F));
  v = fmaxf(v, SWZF(v, 0x101F));
  v = fmaxf(v, SWZF(v, 0x201F));
  return v;
}
__device__ __forceinline__ float redsum16(float v) {
  v += SWZF(v, 0x041F);
  v += SWZF(v, 0x081F);
  v += SWZF(v, 0x101F);
  v += SWZF(v, 0x201F);
  return v;
}

// ---------------------------------------------------------------------------
// One-time f32 -> packed bf16x2 conversion.
// ---------------------------------------------------------------------------
__global__ __launch_bounds__(256) void cvt_f32_bf16_kernel(
    const float* __restrict__ in, unsigned* __restrict__ out, long n2) {
  long i = (long)blockIdx.x * blockDim.x + threadIdx.x;
  if (i < n2) {
    f32x2 v = *(const f32x2*)(in + 2 * i);
    out[i] = pack_bf16(v.x, v.y);
  }
}

// ---------------------------------------------------------------------------
// GEMM: C[M][Nn] = A[M][K] * W[Nn][K]^T + bias[Nn]   (A,W bf16; acc f32).
// Block tile 128x128, 8 waves of 32x64; k-step 64 (16 WMMA / step / wave).
// Double-buffered async LDS staging: 8 async b128 per thread per tile;
// steady state waits s_wait_asynccnt 8 (previous tile landed, next in flight).
// mode 0: f32 C. mode 1: bf16 row-major C. mode 2: bf16 per-head transposed
//         (V-layout: Cb[((n*16+h)*64+hd)*2048 + l], M=8192 tokens, L=2048).
// ---------------------------------------------------------------------------
__global__ __launch_bounds__(256) void gemm_bf16_kernel(
    const unsigned short* __restrict__ A, const unsigned short* __restrict__ W,
    const float* __restrict__ bias, float* __restrict__ Cf,
    unsigned short* __restrict__ Cb, int M, int Nn, int K, int mode) {
  __shared__ __align__(16) unsigned int As[2][128 * 32];  // [row][64 bf16]
  __shared__ __align__(16) unsigned int Bs[2][128 * 32];

  const int t = threadIdx.x;
  const int lane = t & 31, wave = t >> 5;
  const int wm = wave & 3, wn = wave >> 2;  // wave grid 4x2 over 128x128
  const int m0g = blockIdx.y * 128, n0g = blockIdx.x * 128;
  const int nl = lane & 15, half = lane >> 4;

  auto stage = [&](int step, int b) {  // 4+4 x 16B async copies per thread
#pragma unroll
    for (int i = 0; i < 4; ++i) {
      int c = t + 256 * i;  // chunk id in [0,1024)
      int row = c >> 3, seg = c & 7;
      async_cp16(A + (long)(m0g + row) * K + step * 64 + seg * 8,
                 (char*)&As[b][0] + row * 128 + seg * 16);
      async_cp16(W + (long)(n0g + row) * K + step * 64 + seg * 8,
                 (char*)&Bs[b][0] + row * 128 + seg * 16);
    }
  };

  v8f acc[2][4] = {};
  const int NSTEP = K >> 6;  // K=1024 -> 16

  stage(0, 0);
  for (int i = 0; i < NSTEP; ++i) {
    const int cur = i & 1;
    if (i + 1 < NSTEP) {
      stage(i + 1, cur ^ 1);
      ASYNC_WAIT(0x8);  // tile i landed; tile i+1 still in flight
    } else {
      ASYNC_WAIT(0x0);
    }
    __syncthreads();

    const char* Ab = (const char*)&As[cur][0];
    const char* Bb = (const char*)&Bs[cur][0];
    v16bf af[2][2], bfr[4][2];
#pragma unroll
    for (int mt = 0; mt < 2; ++mt)
#pragma unroll
      for (int ks = 0; ks < 2; ++ks)
        af[mt][ks] = frag_a_lds(Ab, 128, wm * 32 + mt * 16 + nl, ks * 64, lane);
#pragma unroll
    for (int nt = 0; nt < 4; ++nt)
#pragma unroll
      for (int ks = 0; ks < 2; ++ks)
        bfr[nt][ks] = frag_b_lds(Bb, 128, wn * 64 + nt * 16 + nl, ks * 64, lane);
#pragma unroll
    for (int mt = 0; mt < 2; ++mt)
#pragma unroll
      for (int nt = 0; nt < 4; ++nt) {
        acc[mt][nt] = wmma_bf16(af[mt][0], bfr[nt][0], acc[mt][nt]);
        acc[mt][nt] = wmma_bf16(af[mt][1], bfr[nt][1], acc[mt][nt]);
      }
    __syncthreads();  // release buf[cur] before it is re-staged at step i+2
  }

  // ---- epilogue (mode hoisted out of the store loops) ----
  if (mode == 0) {
#pragma unroll
    for (int mt = 0; mt < 2; ++mt)
#pragma unroll
      for (int nt = 0; nt < 4; ++nt) {
        int ncol = n0g + wn * 64 + nt * 16 + nl;
        float b = bias[ncol];
#pragma unroll
        for (int r = 0; r < 8; ++r) {
          int mrow = m0g + wm * 32 + mt * 16 + r + half * 8;
          Cf[(long)mrow * Nn + ncol] = acc[mt][nt][r] + b;
        }
      }
  } else if (mode == 1) {
#pragma unroll
    for (int mt = 0; mt < 2; ++mt)
#pragma unroll
      for (int nt = 0; nt < 4; ++nt) {
        int ncol = n0g + wn * 64 + nt * 16 + nl;
        float b = bias[ncol];
#pragma unroll
        for (int r = 0; r < 8; ++r) {
          int mrow = m0g + wm * 32 + mt * 16 + r + half * 8;
          Cb[(long)mrow * Nn + ncol] = f32_to_bf16(acc[mt][nt][r] + b);
        }
      }
  } else {  // V: transposed per head, L=2048, H=16, hd=64
#pragma unroll
    for (int mt = 0; mt < 2; ++mt)
#pragma unroll
      for (int nt = 0; nt < 4; ++nt) {
        int ncol = n0g + wn * 64 + nt * 16 + nl;
        float b = bias[ncol];
        int h = ncol >> 6, hd = ncol & 63;
#pragma unroll
        for (int r = 0; r < 8; ++r) {
          int mrow = m0g + wm * 32 + mt * 16 + r + half * 8;
          int nb = mrow >> 11, l = mrow & 2047;
          Cb[((long)((nb * 16 + h) * 64 + hd)) * 2048 + l] =
              f32_to_bf16(acc[mt][nt][r] + b);
        }
      }
  }
}

// ---------------------------------------------------------------------------
// Flash attention. Grid: (L/256, H, N). Block = 8 waves; wave w owns queries
// [q0blk + 32w, +32). Double-buffered async staging of 32-key K/V^T tiles.
// ---------------------------------------------------------------------------
__global__ __launch_bounds__(256) void attn_kernel(
    const unsigned short* __restrict__ Qb, const unsigned short* __restrict__ Kb,
    const unsigned short* __restrict__ Vtb,
    const unsigned char* __restrict__ amask, unsigned short* __restrict__ Ob,
    int L) {
  const int D = 1024, HD = 64, H = 16;
  __shared__ __align__(16) unsigned int Ks[2][32 * 32];     // [key][64 bf16]
  __shared__ __align__(16) unsigned int Vts[2][64 * 16];    // [hd][32 keys bf16]
  __shared__ __align__(16) unsigned short Ps[8 * 32 * 32];  // per-wave P

  const int t = threadIdx.x, lane = t & 31, wave = t >> 5;
  const int nl = lane & 15, half = lane >> 4;
  const int qb = blockIdx.x, h = blockIdx.y, n = blockIdx.z;
  const int q0blk = qb * 256;
  const int qw0 = q0blk + wave * 32;

  const unsigned short* Qp = Qb + (long)n * L * D + h * HD;
  const unsigned short* Kp = Kb + (long)n * L * D + h * HD;
  const unsigned short* Vtp = Vtb + ((long)(n * H + h) * HD) * L;

  auto stage = [&](int kb2, int b) {  // 1+1 x 16B async copies per thread
    int key = t >> 3, seg = t & 7;
    async_cp16(Kp + (long)(kb2 * 32 + key) * D + seg * 8,
               (char*)&Ks[b][0] + key * 128 + seg * 16);
    int hd = t >> 2, sg2 = t & 3;
    async_cp16(Vtp + (long)hd * L + kb2 * 32 + sg2 * 8,
               (char*)&Vts[b][0] + hd * 64 + sg2 * 16);
  };

  // Q fragments loaded once per wave (hd=64 => 2 K-steps of 32).
  v16bf qfrag[2][2];
#pragma unroll
  for (int mt = 0; mt < 2; ++mt)
#pragma unroll
    for (int ks = 0; ks < 2; ++ks)
      qfrag[mt][ks] = frag_a_global_bf16(Qp, D, qw0 + mt * 16, ks * 32, lane);

  v8f oacc[2][4] = {};
  float rmax[2][8], rsum[2][8];
#pragma unroll
  for (int mt = 0; mt < 2; ++mt)
#pragma unroll
    for (int r = 0; r < 8; ++r) { rmax[mt][r] = -INFINITY; rsum[mt][r] = 0.f; }

  const int kb_max_wave = qw0 >> 5;
  const int kb_max_blk = (q0blk + 255) >> 5;
  const float scale = 0.125f;  // 1/sqrt(64)
  unsigned short* myP = &Ps[wave * 32 * 32];

  stage(0, 0);
  for (int kb = 0; kb <= kb_max_blk; ++kb) {
    const int cur = kb & 1;
    if (kb < kb_max_blk) {
      stage(kb + 1, cur ^ 1);
      ASYNC_WAIT(0x2);  // tile kb landed; tile kb+1 in flight
    } else {
      ASYNC_WAIT(0x0);
    }
    __syncthreads();

    if (kb <= kb_max_wave) {
      // ---- S = Q * K^T  (8 WMMAs) ----
      v16bf kf[2][2];
#pragma unroll
      for (int nt = 0; nt < 2; ++nt)
#pragma unroll
        for (int ks = 0; ks < 2; ++ks)
          kf[nt][ks] = frag_b_lds((const char*)&Ks[cur][0], 128, nt * 16 + nl,
                                  ks * 64, lane);

      v8f s[2][2];
#pragma unroll
      for (int mt = 0; mt < 2; ++mt)
#pragma unroll
        for (int nt = 0; nt < 2; ++nt) {
          v8f a = {};
          a = wmma_bf16(qfrag[mt][0], kf[nt][0], a);
          a = wmma_bf16(qfrag[mt][1], kf[nt][1], a);
          s[mt][nt] = a;
        }

      // ---- scale + causal/key mask ----
#pragma unroll
      for (int nt = 0; nt < 2; ++nt) {
        int kg = kb * 32 + nt * 16 + nl;
        bool amb = amask[(long)n * L + kg] != 0;
#pragma unroll
        for (int mt = 0; mt < 2; ++mt)
#pragma unroll
          for (int r = 0; r < 8; ++r) {
            int qg = qw0 + mt * 16 + r + half * 8;
            float v = s[mt][nt][r] * scale;
            s[mt][nt][r] = (amb && kg <= qg) ? v : -INFINITY;
          }
      }

      // ---- online softmax + P write ----
#pragma unroll
      for (int mt = 0; mt < 2; ++mt) {
        float al[8];
#pragma unroll
        for (int r = 0; r < 8; ++r) {
          float mx = redmax16(fmaxf(s[mt][0][r], s[mt][1][r]));
          float mnew = fmaxf(rmax[mt][r], mx);
          float alpha = __expf(rmax[mt][r] - mnew);
          rmax[mt][r] = mnew;
          float e0 = __expf(s[mt][0][r] - mnew);
          float e1 = __expf(s[mt][1][r] - mnew);
          s[mt][0][r] = e0; s[mt][1][r] = e1;
          rsum[mt][r] = rsum[mt][r] * alpha + redsum16(e0 + e1);
          al[r] = alpha;
        }
#pragma unroll
        for (int nt2 = 0; nt2 < 4; ++nt2)
#pragma unroll
          for (int r = 0; r < 8; ++r) oacc[mt][nt2][r] *= al[r];
#pragma unroll
        for (int nt = 0; nt < 2; ++nt)
#pragma unroll
          for (int r = 0; r < 8; ++r) {
            int mrow = mt * 16 + r + half * 8;
            myP[mrow * 32 + nt * 16 + nl] = f32_to_bf16(s[mt][nt][r]);
          }
      }

      // ---- O += P * V  (8 WMMAs) ----
      v16bf pf[2], vfrag[4];
#pragma unroll
      for (int mt = 0; mt < 2; ++mt)
        pf[mt] = frag_a_lds((const char*)myP, 64, mt * 16 + nl, 0, lane);
#pragma unroll
      for (int nt2 = 0; nt2 < 4; ++nt2)
        vfrag[nt2] =
            frag_b_lds((const char*)&Vts[cur][0], 64, nt2 * 16 + nl, 0, lane);
#pragma unroll
      for (int mt = 0; mt < 2; ++mt)
#pragma unroll
        for (int nt2 = 0; nt2 < 4; ++nt2)
          oacc[mt][nt2] = wmma_bf16(pf[mt], vfrag[nt2], oacc[mt][nt2]);
    }
    __syncthreads();  // release buf[cur] before restage at kb+2
  }

  // ---- epilogue: divide by row sum, store bf16 [n][q][h*64+d] ----
#pragma unroll
  for (int mt = 0; mt < 2; ++mt)
#pragma unroll
    for (int r = 0; r < 8; ++r) {
      float inv = 1.0f / rsum[mt][r];
      int qg = qw0 + mt * 16 + r + half * 8;
#pragma unroll
      for (int nt2 = 0; nt2 < 4; ++nt2) {
        int col = h * HD + nt2 * 16 + nl;
        Ob[((long)n * L + qg) * D + col] = f32_to_bf16(oacc[mt][nt2][r] * inv);
      }
    }
}

// ---------------------------------------------------------------------------
extern "C" void kernel_launch(void* const* d_in, const int* in_sizes, int n_in,
                              void* d_out, int out_size, void* d_ws,
                              size_t ws_size, hipStream_t stream) {
  (void)in_sizes; (void)n_in; (void)out_size; (void)ws_size;
  const int Nb = 4, L = 2048, D = 1024, H = 16;
  const long MT = (long)Nb * L;  // 8192 tokens

  const float* x  = (const float*)d_in[0];
  const float* Wq = (const float*)d_in[1];
  const float* bq = (const float*)d_in[2];
  const float* Wk = (const float*)d_in[3];
  const float* bk = (const float*)d_in[4];
  const float* Wv = (const float*)d_in[5];
  const float* bv = (const float*)d_in[6];
  const float* Wo = (const float*)d_in[7];
  const float* bo = (const float*)d_in[8];
  // d_in[9] = causal mask (recomputed arithmetically in-kernel)
  const unsigned char* amask = (const unsigned char*)d_in[10];
  float* out = (float*)d_out;

  // Workspace layout (bf16 buffers), ~88 MB total.
  char* ws = (char*)d_ws;
  unsigned short* xb  = (unsigned short*)(ws);                        // 16 MB
  unsigned short* Wqb = (unsigned short*)(ws + (16L << 20));          //  2 MB
  unsigned short* Wkb = (unsigned short*)(ws + (18L << 20));
  unsigned short* Wvb = (unsigned short*)(ws + (20L << 20));
  unsigned short* Wob = (unsigned short*)(ws + (22L << 20));
  unsigned short* Qbf = (unsigned short*)(ws + (24L << 20));          // 16 MB
  unsigned short* Kbf = (unsigned short*)(ws + (40L << 20));          // 16 MB
  unsigned short* Vtb = (unsigned short*)(ws + (56L << 20));          // 16 MB
  unsigned short* Abf = (unsigned short*)(ws + (72L << 20));          // 16 MB

  // One-time conversions.
  long nx2 = MT * D / 2, nw2 = (long)D * D / 2;
  cvt_f32_bf16_kernel<<<(unsigned)(nx2 / 256), 256, 0, stream>>>(x, (unsigned*)xb, nx2);
  cvt_f32_bf16_kernel<<<(unsigned)(nw2 / 256), 256, 0, stream>>>(Wq, (unsigned*)Wqb, nw2);
  cvt_f32_bf16_kernel<<<(unsigned)(nw2 / 256), 256, 0, stream>>>(Wk, (unsigned*)Wkb, nw2);
  cvt_f32_bf16_kernel<<<(unsigned)(nw2 / 256), 256, 0, stream>>>(Wv, (unsigned*)Wvb, nw2);
  cvt_f32_bf16_kernel<<<(unsigned)(nw2 / 256), 256, 0, stream>>>(Wo, (unsigned*)Wob, nw2);

  dim3 gg(D / 128, (unsigned)(MT / 128));  // (8, 64)
  gemm_bf16_kernel<<<gg, 256, 0, stream>>>(xb, Wqb, bq, nullptr, Qbf, (int)MT, D, D, 1);
  gemm_bf16_kernel<<<gg, 256, 0, stream>>>(xb, Wkb, bk, nullptr, Kbf, (int)MT, D, D, 1);
  gemm_bf16_kernel<<<gg, 256, 0, stream>>>(xb, Wvb, bv, nullptr, Vtb, (int)MT, D, D, 2);

  attn_kernel<<<dim3(L / 256, H, Nb), 256, 0, stream>>>(Qbf, Kbf, Vtb, amask, Abf, L);

  gemm_bf16_kernel<<<gg, 256, 0, stream>>>(Abf, Wob, bo, out, nullptr, (int)MT, D, D, 0);
}
